// SelfAttention_77180562309122
// MI455X (gfx1250) — matrix operand order repaired and checked by reading, not compiled, not run on hardware
//
#include <hip/hip_runtime.h>
#include <hip/hip_bf16.h>

// Self-attention, B=4, T=4096, D=768, H=64, causal, fp32 I/O.
// (1) weight transpose->f16, (2) fused QKV projection (v_wmma_f32_16x16x32_f16),
// (3) flash attention: cooperative async K/V staging to LDS (ASYNCcnt),
//     online softmax, WMMA for QK^T and P*V.

typedef __attribute__((ext_vector_type(16))) _Float16 v16h;
typedef __attribute__((ext_vector_type(8)))  _Float16 v8h;
typedef __attribute__((ext_vector_type(8)))  float    v8f;

union V16 { v16h v; v8h h[2]; };

#define T_SEQ   4096
#define BATCH   4
#define NTOK    16384      // BATCH * T_SEQ
#define DMODEL  768
#define HSZ     64
#define KPAD    72         // 64 + 8 halves pad -> conflict-free 16-row reads
#define VPAD    40         // 32 + 8 halves pad
#define AW      4          // waves per attention block

static __device__ __forceinline__ v8f wmma16(v16h a, v16h b, v8f c) {
  return __builtin_amdgcn_wmma_f32_16x16x32_f16(false, a, false, b, (short)0, c,
                                                false, false);
}

// CDNA5 async copy: global -> LDS, 16B per lane, tracked by ASYNCcnt.
static __device__ __forceinline__ void async_b128(const _Float16* g,
                                                  _Float16* lds) {
  unsigned loff = (unsigned)(unsigned long long)lds;  // LDS addr = addr[31:0]
  asm volatile("global_load_async_to_lds_b128 %0, %1, off"
               :: "v"(loff), "v"(g) : "memory");
}
static __device__ __forceinline__ void wait_async0() {
  asm volatile("s_wait_asynccnt 0x0" ::: "memory");
}

// ---------------------------------------------------------------------------
// Kernel 1: Wt[w][n][k] = (f16) W_w[k][n]
// ---------------------------------------------------------------------------
__global__ void prep_w_kernel(const float* __restrict__ Wq,
                              const float* __restrict__ Wk,
                              const float* __restrict__ Wv,
                              _Float16* __restrict__ Wt) {
  int i = blockIdx.x * 256 + threadIdx.x;
  if (i >= 3 * HSZ * DMODEL) return;
  int w   = i / (HSZ * DMODEL);
  int rem = i - w * (HSZ * DMODEL);
  int n   = rem / DMODEL;
  int k   = rem - n * DMODEL;
  const float* W = (w == 0) ? Wq : ((w == 1) ? Wk : Wv);
  Wt[i] = (_Float16)W[k * HSZ + n];
}

// ---------------------------------------------------------------------------
// Kernel 2: fused QKV projection. One wave -> 16 tokens x 64 cols for Q,K,V.
// B fragments loaded in groups of 6 so loads overlap WMMAs.
// ---------------------------------------------------------------------------
__global__ __launch_bounds__(256) void proj_kernel(
    const float* __restrict__ x, const _Float16* __restrict__ Wt,
    _Float16* __restrict__ Qh, _Float16* __restrict__ Kh,
    _Float16* __restrict__ Vt) {
  const int wave = threadIdx.x >> 5;
  const int lid  = threadIdx.x & 31;
  const int lh   = lid >> 4;
  const int ln   = lid & 15;
  const int rt   = blockIdx.x * 8 + wave;   // 0..1023
  const int r0   = rt << 4;

  v8f accq[4] = {{}, {}, {}, {}};
  v8f acck[4] = {{}, {}, {}, {}};
  v8f accv[4] = {{}, {}, {}, {}};

  const float* xrow = x + (size_t)(r0 + ln) * DMODEL;

  for (int kk = 0; kk < DMODEL; kk += 32) {
    const int base = kk + 16 * lh;
    float4 f0 = *(const float4*)(xrow + base);
    float4 f1 = *(const float4*)(xrow + base + 4);
    float4 f2 = *(const float4*)(xrow + base + 8);
    float4 f3 = *(const float4*)(xrow + base + 12);
    V16 A;
    A.v[0]  = (_Float16)f0.x; A.v[1]  = (_Float16)f0.y;
    A.v[2]  = (_Float16)f0.z; A.v[3]  = (_Float16)f0.w;
    A.v[4]  = (_Float16)f1.x; A.v[5]  = (_Float16)f1.y;
    A.v[6]  = (_Float16)f1.z; A.v[7]  = (_Float16)f1.w;
    A.v[8]  = (_Float16)f2.x; A.v[9]  = (_Float16)f2.y;
    A.v[10] = (_Float16)f2.z; A.v[11] = (_Float16)f2.w;
    A.v[12] = (_Float16)f3.x; A.v[13] = (_Float16)f3.y;
    A.v[14] = (_Float16)f3.z; A.v[15] = (_Float16)f3.w;

    const int kbp = kk + 16 * lh;
#pragma unroll
    for (int grp = 0; grp < 2; ++grp) {   // two ns per group: 6 B tiles live
      V16 Bt[6];
#pragma unroll
      for (int j = 0; j < 2; ++j) {
        const int ns = grp * 2 + j;
        const int n  = ns * 16 + ln;
        const _Float16* w0 = Wt + (size_t)n * DMODEL + kbp;
        Bt[j * 3 + 0].h[0] = ((const v8h*)w0)[0];
        Bt[j * 3 + 0].h[1] = ((const v8h*)w0)[1];
        const _Float16* w1 = w0 + HSZ * DMODEL;
        Bt[j * 3 + 1].h[0] = ((const v8h*)w1)[0];
        Bt[j * 3 + 1].h[1] = ((const v8h*)w1)[1];
        const _Float16* w2 = w1 + HSZ * DMODEL;
        Bt[j * 3 + 2].h[0] = ((const v8h*)w2)[0];
        Bt[j * 3 + 2].h[1] = ((const v8h*)w2)[1];
      }
#pragma unroll
      for (int j = 0; j < 2; ++j) {
        const int ns = grp * 2 + j;
        accq[ns] = wmma16(A.v, Bt[j * 3 + 0].v, accq[ns]);
        acck[ns] = wmma16(A.v, Bt[j * 3 + 1].v, acck[ns]);
        accv[ns] = wmma16(A.v, Bt[j * 3 + 2].v, accv[ns]);
      }
    }
  }

#pragma unroll
  for (int ns = 0; ns < 4; ++ns) {
    const int n = ns * 16 + ln;
#pragma unroll
    for (int r = 0; r < 8; ++r) {
      const int tok = r0 + r + 8 * lh;
      Qh[(size_t)tok * HSZ + n] = (_Float16)(accq[ns][r] * 0.125f);
      Kh[(size_t)tok * HSZ + n] = (_Float16)(acck[ns][r]);
    }
    v8h pv;
#pragma unroll
    for (int r = 0; r < 8; ++r) pv[r] = (_Float16)accv[ns][r];
    *(v8h*)(Vt + (size_t)n * NTOK + r0 + 8 * lh) = pv;
  }
}

// ---------------------------------------------------------------------------
// Kernel 3: flash attention, 4 waves = 64 query rows per block, cooperative
// double-buffered async K/V staging to LDS.
// ---------------------------------------------------------------------------
__global__ __launch_bounds__(32 * AW) void attn_kernel(
    const _Float16* __restrict__ Qh, const _Float16* __restrict__ Kh,
    const _Float16* __restrict__ Vt, float* __restrict__ out) {
  __shared__ __align__(16) _Float16 Klds[2][32][KPAD];
  __shared__ __align__(16) _Float16 Vlds[2][HSZ][VPAD];
  __shared__ __align__(16) _Float16 Plds[AW][16][VPAD];

  const int tid  = threadIdx.x;
  const int wave = tid >> 5;
  const int lid  = tid & 31;
  const int lh   = lid >> 4;
  const int ln   = lid & 15;
  const int g    = blockIdx.x * AW + wave;   // query tile id (0..1023)
  const int b    = g >> 8;
  const int t0   = (g & 255) << 4;           // first query row (in-batch)
  const int gb   = b << 12;                  // batch token base
  const int kend = ((blockIdx.x * AW + (AW - 1)) & 255) << 4;  // block bound

  // Q A-fragments (h-split 0..31 / 32..63)
  const _Float16* qrow = Qh + (size_t)(gb + t0 + ln) * HSZ;
  V16 Aq0, Aq1;
  Aq0.h[0] = *(const v8h*)(qrow + 16 * lh);
  Aq0.h[1] = *(const v8h*)(qrow + 16 * lh + 8);
  Aq1.h[0] = *(const v8h*)(qrow + 32 + 16 * lh);
  Aq1.h[1] = *(const v8h*)(qrow + 32 + 16 * lh + 8);

  v8f O0 = {}, O1 = {}, O2 = {}, O3 = {};
  float m[8], l[8];
#pragma unroll
  for (int r = 0; r < 8; ++r) { m[r] = -1e30f; l[r] = 0.0f; }

  // prologue: stage key block 0 into buffer 0 (all 128 threads, 4x16B each)
#pragma unroll
  for (int i = 0; i < 2; ++i) {
    const int idx = tid + i * 128;
    const int key = idx >> 3, kp = idx & 7;
    async_b128(Kh + (size_t)(gb + key) * HSZ + kp * 8, &Klds[0][key][kp * 8]);
    const int h = idx >> 2, vp = idx & 3;
    async_b128(Vt + (size_t)h * NTOK + gb + vp * 8, &Vlds[0][h][vp * 8]);
  }

  for (int kb = 0; kb <= kend; kb += 32) {
    const int cur = (kb >> 5) & 1;
    wait_async0();      // this wave's staging writes have landed in LDS
    __syncthreads();    // publish buffer `cur`; prior reads of `cur^1` done

    if (kb + 32 <= kend) {   // stage next block into the other buffer
      const int nb = cur ^ 1;
#pragma unroll
      for (int i = 0; i < 2; ++i) {
        const int idx = tid + i * 128;
        const int key = idx >> 3, kp = idx & 7;
        async_b128(Kh + (size_t)(gb + kb + 32 + key) * HSZ + kp * 8,
                   &Klds[nb][key][kp * 8]);
        const int h = idx >> 2, vp = idx & 3;
        async_b128(Vt + (size_t)h * NTOK + gb + kb + 32 + vp * 8,
                   &Vlds[nb][h][vp * 8]);
      }
    }

    if (kb <= t0) {   // this wave still inside its causal range
      // ---- K fragments: 4 tiles (2 key sub-tiles x 2 h-splits), all live ----
      V16 BK[4];
#pragma unroll
      for (int st = 0; st < 2; ++st) {
        const _Float16* kr = &Klds[cur][st * 16 + ln][0];
        BK[st * 2 + 0].h[0] = *(const v8h*)(kr + 16 * lh);
        BK[st * 2 + 0].h[1] = *(const v8h*)(kr + 16 * lh + 8);
        BK[st * 2 + 1].h[0] = *(const v8h*)(kr + 32 + 16 * lh);
        BK[st * 2 + 1].h[1] = *(const v8h*)(kr + 32 + 16 * lh + 8);
      }
      // ---- V fragments prefetched now; latency hides under softmax ----
      V16 BV[4];
#pragma unroll
      for (int hs = 0; hs < 4; ++hs) {
        const _Float16* vr = &Vlds[cur][hs * 16 + ln][16 * lh];
        BV[hs].h[0] = *(const v8h*)(vr);
        BV[hs].h[1] = *(const v8h*)(vr + 8);
      }

      // ---- S = Q K^T ----
      v8f S[2];
#pragma unroll
      for (int st = 0; st < 2; ++st) {
        v8f s = {};
        s = wmma16(Aq0.v, BK[st * 2 + 0].v, s);
        s = wmma16(Aq1.v, BK[st * 2 + 1].v, s);
        S[st] = s;
      }

      // ---- causal mask on ragged blocks ----
      if (kb + 31 > t0) {
#pragma unroll
        for (int st = 0; st < 2; ++st) {
          const int key = kb + st * 16 + ln;
#pragma unroll
          for (int r = 0; r < 8; ++r)
            if (key > t0 + r + 8 * lh) S[st][r] = -1e30f;
        }
      }

      // ---- online softmax (rows live across 16 lanes of a half) ----
      float scale[8];
#pragma unroll
      for (int r = 0; r < 8; ++r) {
        float mx = fmaxf(S[0][r], S[1][r]);
        mx = fmaxf(mx, __shfl_xor(mx, 1, 32));
        mx = fmaxf(mx, __shfl_xor(mx, 2, 32));
        mx = fmaxf(mx, __shfl_xor(mx, 4, 32));
        mx = fmaxf(mx, __shfl_xor(mx, 8, 32));
        const float mn = fmaxf(m[r], mx);
        const float p0 = __expf(S[0][r] - mn);
        const float p1 = __expf(S[1][r] - mn);
        S[0][r] = p0; S[1][r] = p1;
        float rs = p0 + p1;
        rs += __shfl_xor(rs, 1, 32);
        rs += __shfl_xor(rs, 2, 32);
        rs += __shfl_xor(rs, 4, 32);
        rs += __shfl_xor(rs, 8, 32);
        scale[r] = __expf(m[r] - mn);
        l[r] = l[r] * scale[r] + rs;
        m[r] = mn;
      }
#pragma unroll
      for (int r = 0; r < 8; ++r) {
        O0[r] *= scale[r]; O1[r] *= scale[r];
        O2[r] *= scale[r]; O3[r] *= scale[r];
      }

      // ---- P: D-layout -> A-layout via wave-private LDS ----
#pragma unroll
      for (int st = 0; st < 2; ++st)
#pragma unroll
        for (int r = 0; r < 8; ++r)
          Plds[wave][r + 8 * lh][st * 16 + ln] = (_Float16)S[st][r];
      const _Float16* prow = &Plds[wave][ln][0];
      V16 Pa;
      Pa.h[0] = *(const v8h*)(prow + 16 * lh);
      Pa.h[1] = *(const v8h*)(prow + 16 * lh + 8);

      // ---- O += P * V ----
      O0 = wmma16(Pa.v, BV[0].v, O0);
      O1 = wmma16(Pa.v, BV[1].v, O1);
      O2 = wmma16(Pa.v, BV[2].v, O2);
      O3 = wmma16(Pa.v, BV[3].v, O3);
    }
  }

  // ---- epilogue: O / l, fp32 store ----
  v8f OO[4] = {O0, O1, O2, O3};
#pragma unroll
  for (int hs = 0; hs < 4; ++hs) {
    const int h = hs * 16 + ln;
#pragma unroll
    for (int r = 0; r < 8; ++r) {
      const int tok = gb + t0 + r + 8 * lh;
      out[(size_t)tok * HSZ + h] = OO[hs][r] / l[r];
    }
  }
}

// ---------------------------------------------------------------------------
extern "C" void kernel_launch(void* const* d_in, const int* in_sizes, int n_in,
                              void* d_out, int out_size, void* d_ws,
                              size_t ws_size, hipStream_t stream) {
  const float* x  = (const float*)d_in[0];
  const float* Wq = (const float*)d_in[1];
  const float* Wk = (const float*)d_in[2];
  const float* Wv = (const float*)d_in[3];
  float* out = (float*)d_out;

  _Float16* ws = (_Float16*)d_ws;
  _Float16* Qh = ws;
  _Float16* Kh = ws + (size_t)NTOK * HSZ;
  _Float16* Vt = ws + (size_t)2 * NTOK * HSZ;
  _Float16* Wt = ws + (size_t)3 * NTOK * HSZ;

  const int wtot = 3 * HSZ * DMODEL;
  prep_w_kernel<<<(wtot + 255) / 256, 256, 0, stream>>>(Wq, Wk, Wv, Wt);
  proj_kernel<<<128, 256, 0, stream>>>(x, Wt, Qh, Kh, Vt);
  attn_kernel<<<(NTOK / 16) / AW, 32 * AW, 0, stream>>>(Qh, Kh, Vt, out);
}
